// QwenGroupedQueryAttention_19250043421023
// MI455X (gfx1250) — matrix operand order repaired and checked by reading, not compiled, not run on hardware
//
#include <hip/hip_runtime.h>
#include <hip/hip_bf16.h>

typedef __bf16 bf16_t;
typedef __attribute__((ext_vector_type(16))) __bf16 v16bf;
typedef __attribute__((ext_vector_type(8)))  float  v8f;

#define B_   2
#define T_   2048
#define E_   2048
#define NH_  16
#define NKV_ 4
#define HD_  128
#define KVD_ 512
#define HALF_ 64

#define WMMA_BF16(a, b, c) \
  __builtin_amdgcn_wmma_f32_16x16x32_bf16(false, (a), false, (b), (short)0, (c), false, false)

// ---------- CDNA5 async global->LDS (ASYNCcnt path) via inline asm ----------
// ISA 15.18.3: GLOBAL_LOAD_ASYNC_TO_LDS_B128, GV mode:
//   each lane copies 16 bytes from its global address to its LDS address
// Generic LDS pointers carry the LDS offset in their low 32 bits (ISA 10.2).
static __device__ __forceinline__ void async_copy_b128(const void* gsrc, void* ldst) {
  asm volatile("global_load_async_to_lds_b128 %0, %1, off"
               :
               : "v"((unsigned int)(size_t)ldst), "v"((unsigned long long)(size_t)gsrc)
               : "memory");
}

template <int N>
static __device__ __forceinline__ void wait_async_le() {
#if __has_builtin(__builtin_amdgcn_s_wait_asynccnt)
  __builtin_amdgcn_s_wait_asynccnt(N);
#else
  asm volatile("s_wait_asynccnt %0" ::"n"(N) : "memory");
#endif
}

// A fragment (16x32, row-major MxK source): lane<16 -> row=lane, K 0..7 & 16..23
//                                           lane>=16 -> row=lane-16, K 8..15 & 24..31
static __device__ __forceinline__ v16bf load_frag_a_g(const bf16_t* base, int stride, int lane) {
  const int r  = lane & 15;
  const int ko = (lane >> 4) << 3;  // 0 or 8
  const bf16_t* p = base + (size_t)r * stride + ko;
  v16bf v;
  ((uint4*)&v)[0] = *(const uint4*)(p);
  ((uint4*)&v)[1] = *(const uint4*)(p + 16);
  return v;
}

// B fragment (32x16) from row-major (N,K) source (i.e. B = W^T):
// lane<16 -> col n=lane, K 0..7 & 16..23; lane>=16 -> col n=lane-16, K 8..15 & 24..31
static __device__ __forceinline__ v16bf load_frag_b_g(const bf16_t* base, int stride, int lane) {
  const int n  = lane & 15;
  const int ko = (lane >> 4) << 3;  // 0 or 8
  const bf16_t* p = base + (size_t)n * stride + ko;
  v16bf v;
  ((uint4*)&v)[0] = *(const uint4*)(p);
  ((uint4*)&v)[1] = *(const uint4*)(p + 16);
  return v;
}

__global__ __launch_bounds__(256) void f32_to_bf16_kernel(const float* __restrict__ in,
                                                          bf16_t* __restrict__ out, int n) {
  int i = blockIdx.x * blockDim.x + threadIdx.x;
  if (i < n) out[i] = (bf16_t)in[i];
}

// C(M,N) = A(M,K)bf16 @ W(N,K)^T bf16 + bias, fp32 accumulate/out.
// Block: 8 waves, 128x128 tile. A/B k-tiles (128x32) staged in LDS with
// double-buffered async global->LDS copies; waves consume via ds_load_b128.
__global__ __launch_bounds__(256) void gemm_bf16_wmma_kernel(
    const bf16_t* __restrict__ A, const bf16_t* __restrict__ W,
    const float* __restrict__ bias, float* __restrict__ C, int M, int N, int K) {
  __shared__ bf16_t As[2][128 * 32];
  __shared__ bf16_t Bs[2][128 * 32];
  const int tid    = threadIdx.x;
  const int lane   = tid & 31;
  const int wave   = tid >> 5;
  const int mBlock = blockIdx.y * 128;
  const int nBlock = blockIdx.x * 128;
  const int mWave  = (wave & 1) * 64;   // LDS-relative
  const int nWave  = (wave >> 1) * 32;  // LDS-relative

  // Stage one 128x32 A tile + one 128x32 B tile (8 KB each = 512 b128 chunks each).
  // Thread handles 2 chunks of each; 4 async instructions per wave per stage.
  auto stage = [&](int buf, int k0) {
    const bf16_t* Ag = A + (size_t)mBlock * K + k0;
    const bf16_t* Wg = W + (size_t)nBlock * K + k0;
#pragma unroll
    for (int j = 0; j < 2; ++j) {
      int c   = tid + j * 256;
      int row = c >> 2;
      int cc  = (c & 3) * 8;
      async_copy_b128(Ag + (size_t)row * K + cc, &As[buf][row * 32 + cc]);
      async_copy_b128(Wg + (size_t)row * K + cc, &Bs[buf][row * 32 + cc]);
    }
  };

  v8f acc[4][2] = {};
  stage(0, 0);
  const int steps = K / 32;
  for (int kstep = 0; kstep < steps; ++kstep) {
    const int buf = kstep & 1;
    if (kstep + 1 < steps) {
      stage(buf ^ 1, (kstep + 1) * 32);
      wait_async_le<4>();  // newest stage (4 async ops) stays in flight
    } else {
      wait_async_le<0>();
    }
    __syncthreads();
    v16bf bf0 = load_frag_b_g(&Bs[buf][nWave * 32], 32, lane);
    v16bf bf1 = load_frag_b_g(&Bs[buf][(nWave + 16) * 32], 32, lane);
#pragma unroll
    for (int mi = 0; mi < 4; ++mi) {
      v16bf af = load_frag_a_g(&As[buf][(mWave + mi * 16) * 32], 32, lane);
      acc[mi][0] = WMMA_BF16(af, bf0, acc[mi][0]);
      acc[mi][1] = WMMA_BF16(af, bf1, acc[mi][1]);
    }
    __syncthreads();  // protect buffer reuse two steps later
  }

  const int col    = lane & 15;
  const int rowOff = (lane >> 4) << 3;
#pragma unroll
  for (int ni = 0; ni < 2; ++ni) {
    int n = nBlock + nWave + ni * 16 + col;
    float bv = bias ? bias[n] : 0.f;
#pragma unroll
    for (int mi = 0; mi < 4; ++mi)
#pragma unroll
      for (int r = 0; r < 8; ++r)
        C[(size_t)(mBlock + mWave + mi * 16 + rowOff + r) * N + n] = acc[mi][ni][r] + bv;
  }
}

// RoPE (half rotation) fp32 -> bf16. Layout (B,T,nHeads,HD_).
__global__ __launch_bounds__(256) void rope_bf16_kernel(
    const float* __restrict__ in, const float* __restrict__ cosT,
    const float* __restrict__ sinT, bf16_t* __restrict__ out, int nHeads, int total) {
  int i = blockIdx.x * blockDim.x + threadIdx.x;
  if (i >= total) return;
  int d    = i & 63;
  int rest = i >> 6;
  int hh   = rest % nHeads;
  int bt   = rest / nHeads;
  int t    = bt % T_;
  const float* pr = in + ((size_t)bt * nHeads + hh) * HD_;
  float x1 = pr[d], x2 = pr[d + HALF_];
  float c = cosT[t * HALF_ + d];
  float s = sinT[t * HALF_ + d];
  bf16_t* po = out + ((size_t)bt * nHeads + hh) * HD_;
  po[d]         = (bf16_t)(x1 * c - x2 * s);
  po[d + HALF_] = (bf16_t)(x1 * s + x2 * c);
}

// V (B,T,NKV,HD) fp32 -> vT (B,NKV,HD,T) bf16
__global__ __launch_bounds__(256) void v_transpose_kernel(const float* __restrict__ v,
                                                          bf16_t* __restrict__ vT, int total) {
  int i = blockIdx.x * blockDim.x + threadIdx.x;
  if (i >= total) return;
  int d    = i & 127;
  int rest = i >> 7;
  int hh   = rest & 3;
  int bt   = rest >> 2;
  int t    = bt % T_;
  int b    = bt / T_;
  vT[(((size_t)b * NKV_ + hh) * HD_ + d) * T_ + t] = (bf16_t)v[i];
}

// Flash attention: one wave = 16 query rows of one (b, head).
// q: (B,T,NH,HD) bf16, k: (B,T,NKV,HD) bf16, vT: (B,NKV,HD,T) bf16, y: (B,T,E) bf16
__global__ __launch_bounds__(256) void flash_attn_kernel(
    const bf16_t* __restrict__ q, const bf16_t* __restrict__ k,
    const bf16_t* __restrict__ vT, bf16_t* __restrict__ y) {
  __shared__ bf16_t ldsP[8][16 * 32];
  const int lane  = threadIdx.x & 31;
  const int wave  = threadIdx.x >> 5;
  const int task  = blockIdx.x * 8 + wave;
  const int qt    = task & 127;        // T/16 = 128 row-tiles
  const int h     = (task >> 7) & 15;  // query head
  const int b     = task >> 11;
  const int qBase = qt << 4;
  const int hk    = h >> 2;            // kv head = h / KV_GROUP
  const float scale = 0.08838834764831845f;  // 1/sqrt(128)

  const bf16_t* qrow = q + ((size_t)b * T_ + qBase) * E_ + h * HD_;
  const bf16_t* krow = k + (size_t)b * T_ * KVD_ + hk * HD_;
  const bf16_t* vtp  = vT + ((size_t)(b * NKV_ + hk)) * HD_ * T_;

  v16bf qf[4];
#pragma unroll
  for (int c = 0; c < 4; ++c) qf[c] = load_frag_a_g(qrow + c * 32, E_, lane);

  float m[8], l[8];
  v8f o[8];
#pragma unroll
  for (int r = 0; r < 8; ++r) { m[r] = -1e30f; l[r] = 0.f; }
#pragma unroll
  for (int dt = 0; dt < 8; ++dt)
#pragma unroll
    for (int r = 0; r < 8; ++r) o[dt][r] = 0.f;

  const int col    = lane & 15;
  const int rowOff = (lane >> 4) << 3;
  const int sEnd   = qBase + 16;

  for (int s = 0; s < sEnd; s += 32) {
    if (s + 32 < sEnd) {  // prefetch next K/V tiles (global_prefetch_b8)
      __builtin_prefetch(krow + (size_t)(s + 32) * KVD_ + lane * 64, 0, 0);
      __builtin_prefetch(vtp + (size_t)lane * 4 * T_ + s + 32, 0, 0);
    }
    v8f sc0 = {}; v8f sc1 = {};
#pragma unroll
    for (int c = 0; c < 4; ++c) {
      v16bf kb0 = load_frag_b_g(krow + (size_t)s * KVD_ + c * 32, KVD_, lane);
      v16bf kb1 = load_frag_b_g(krow + (size_t)(s + 16) * KVD_ + c * 32, KVD_, lane);
      sc0 = WMMA_BF16(qf[c], kb0, sc0);
      sc1 = WMMA_BF16(qf[c], kb1, sc1);
    }
    // online softmax: each score row lives in one 16-lane half (cols across lanes)
#pragma unroll
    for (int r = 0; r < 8; ++r) {
      int row  = qBase + rowOff + r;
      float v0 = sc0[r] * scale;
      float v1 = sc1[r] * scale;
      if (s + col > row)      v0 = -1e30f;   // causal mask
      if (s + 16 + col > row) v1 = -1e30f;
      float rowmax = fmaxf(v0, v1);
      rowmax = fmaxf(rowmax, __shfl_xor(rowmax, 1, 32));
      rowmax = fmaxf(rowmax, __shfl_xor(rowmax, 2, 32));
      rowmax = fmaxf(rowmax, __shfl_xor(rowmax, 4, 32));
      rowmax = fmaxf(rowmax, __shfl_xor(rowmax, 8, 32));
      float mnew  = fmaxf(m[r], rowmax);
      float alpha = __expf(m[r] - mnew);
      m[r] = mnew;
      float p0 = __expf(v0 - mnew);
      float p1 = __expf(v1 - mnew);
      float rs = p0 + p1;
      rs += __shfl_xor(rs, 1, 32);
      rs += __shfl_xor(rs, 2, 32);
      rs += __shfl_xor(rs, 4, 32);
      rs += __shfl_xor(rs, 8, 32);
      l[r] = l[r] * alpha + rs;
#pragma unroll
      for (int dt = 0; dt < 8; ++dt) o[dt][r] *= alpha;
      sc0[r] = p0;
      sc1[r] = p1;
    }
    // P (C-layout) -> A-fragment layout through per-wave LDS slab (in-order per wave)
    bf16_t* slab = &ldsP[wave][0];
#pragma unroll
    for (int r = 0; r < 8; ++r) {
      slab[(rowOff + r) * 32 + col]      = (bf16_t)sc0[r];
      slab[(rowOff + r) * 32 + 16 + col] = (bf16_t)sc1[r];
    }
    v16bf pf;
    {
      int rr = lane & 15;
      int ko = (lane >> 4) << 3;
      const bf16_t* p = slab + rr * 32 + ko;
      ((uint4*)&pf)[0] = *(const uint4*)(p);
      ((uint4*)&pf)[1] = *(const uint4*)(p + 16);
    }
    // O += P @ V  (B fragment contiguous thanks to transposed V)
#pragma unroll
    for (int dt = 0; dt < 8; ++dt) {
      v16bf vb = load_frag_b_g(vtp + (size_t)(dt * 16) * T_ + s, T_, lane);
      o[dt] = WMMA_BF16(pf, vb, o[dt]);
    }
  }

  // normalize + store y (b, t, h*HD + d) bf16
#pragma unroll
  for (int r = 0; r < 8; ++r) {
    float inv = 1.f / l[r];
#pragma unroll
    for (int dt = 0; dt < 8; ++dt)
      y[((size_t)b * T_ + qBase + rowOff + r) * E_ + h * HD_ + dt * 16 + col] =
          (bf16_t)(o[dt][r] * inv);
  }
}

extern "C" void kernel_launch(void* const* d_in, const int* in_sizes, int n_in,
                              void* d_out, int out_size, void* d_ws, size_t ws_size,
                              hipStream_t stream) {
  (void)in_sizes; (void)n_in; (void)out_size; (void)ws_size;
  const float* x  = (const float*)d_in[0];
  const float* Wq = (const float*)d_in[1];
  const float* bq = (const float*)d_in[2];
  const float* Wk = (const float*)d_in[3];
  const float* bk = (const float*)d_in[4];
  const float* Wv = (const float*)d_in[5];
  const float* bv = (const float*)d_in[6];
  const float* Wo = (const float*)d_in[7];
  const float* fc = (const float*)d_in[8];
  const float* fs = (const float*)d_in[9];

  char* ws = (char*)d_ws;
  size_t off = 0;
  auto alloc = [&](size_t bytes) {
    char* p = ws + off;
    off += (bytes + 255) & ~(size_t)255;
    return p;
  };
  const size_t nX  = (size_t)B_ * T_ * E_;    // 8,388,608
  const size_t nKV = (size_t)B_ * T_ * KVD_;  // 2,097,152

  bf16_t* xb  = (bf16_t*)alloc(nX * 2);
  bf16_t* Wqb = (bf16_t*)alloc((size_t)E_ * E_ * 2);
  bf16_t* Wkb = (bf16_t*)alloc((size_t)KVD_ * E_ * 2);
  bf16_t* Wvb = (bf16_t*)alloc((size_t)KVD_ * E_ * 2);
  bf16_t* Wob = (bf16_t*)alloc((size_t)E_ * E_ * 2);
  float*  Qf  = (float*)alloc(nX * 4);
  float*  Kf  = (float*)alloc(nKV * 4);
  float*  Vf  = (float*)alloc(nKV * 4);
  bf16_t* qb  = (bf16_t*)alloc(nX * 2);
  bf16_t* kb  = (bf16_t*)alloc(nKV * 2);
  bf16_t* vTb = (bf16_t*)alloc(nKV * 2);
  bf16_t* yb  = (bf16_t*)alloc(nX * 2);

  auto cvt = [&](const float* in, bf16_t* out, size_t n) {
    int nn = (int)n;
    f32_to_bf16_kernel<<<(nn + 255) / 256, 256, 0, stream>>>(in, out, nn);
  };
  cvt(x, xb, nX);
  cvt(Wq, Wqb, (size_t)E_ * E_);
  cvt(Wk, Wkb, (size_t)KVD_ * E_);
  cvt(Wv, Wvb, (size_t)KVD_ * E_);
  cvt(Wo, Wob, (size_t)E_ * E_);

  const int M = B_ * T_;  // 4096
  gemm_bf16_wmma_kernel<<<dim3(E_ / 128, M / 128), 256, 0, stream>>>(xb, Wqb, bq, Qf, M, E_, E_);
  gemm_bf16_wmma_kernel<<<dim3(KVD_ / 128, M / 128), 256, 0, stream>>>(xb, Wkb, bk, Kf, M, KVD_, E_);
  gemm_bf16_wmma_kernel<<<dim3(KVD_ / 128, M / 128), 256, 0, stream>>>(xb, Wvb, bv, Vf, M, KVD_, E_);

  int totQ = B_ * T_ * NH_ * HALF_;
  rope_bf16_kernel<<<(totQ + 255) / 256, 256, 0, stream>>>(Qf, fc, fs, qb, NH_, totQ);
  int totK = B_ * T_ * NKV_ * HALF_;
  rope_bf16_kernel<<<(totK + 255) / 256, 256, 0, stream>>>(Kf, fc, fs, kb, NKV_, totK);
  int totV = (int)nKV;
  v_transpose_kernel<<<(totV + 255) / 256, 256, 0, stream>>>(Vf, vTb, totV);

  flash_attn_kernel<<<(B_ * NH_ * (T_ / 16)) / 8, 256, 0, stream>>>(qb, kb, vTb, yb);

  gemm_bf16_wmma_kernel<<<dim3(E_ / 128, M / 128), 256, 0, stream>>>(yb, Wob, nullptr,
                                                                     (float*)d_out, M, E_, E_);
}
